// Cluster_17360257810848
// MI455X (gfx1250) — matrix operand order repaired and checked by reading, not compiled, not run on hardware
//
#include <hip/hip_runtime.h>

// ---------------------------------------------------------------------------
// CDNA5 (gfx1250) implementation of the 3D cluster-attention block.
// - bf16 WMMA (v_wmma_f32_16x16x32_bf16) with f32 accumulation for the three
//   96x96 channel GEMMs; f32 everywhere else.
// - Async global->LDS staging (GLOBAL_LOAD_ASYNC_TO_LDS_B32/B128) when the
//   toolchain exposes the builtins; synchronous fallback otherwise.
// ---------------------------------------------------------------------------

typedef __attribute__((ext_vector_type(16))) __bf16 v16bf;
typedef __attribute__((ext_vector_type(8)))  float  v8f;
typedef __attribute__((ext_vector_type(4)))  float  v4f;
typedef __attribute__((ext_vector_type(4)))  int    v4i;

#define AST 100   // LDS tile row stride in floats (96 + 4 pad): 400B rows,
                  // 16B-aligned chunks, 100%64=36 -> no bank conflicts over 16 rows

#if defined(__has_builtin)
#  if __has_builtin(__builtin_amdgcn_global_load_async_to_lds_b32) && \
      __has_builtin(__builtin_amdgcn_global_load_async_to_lds_b128)
#    define USE_ASYNC_LDS 1
#  endif
#endif
#ifndef USE_ASYNC_LDS
#  define USE_ASYNC_LDS 0
#endif

#if USE_ASYNC_LDS
__device__ __forceinline__ void async_copy_b32(const float* g, float* l) {
  __builtin_amdgcn_global_load_async_to_lds_b32(
      (__attribute__((address_space(1))) int*)g,
      (__attribute__((address_space(3))) int*)l, 0, 0);
}
__device__ __forceinline__ void async_copy_b128(const float* g, float* l) {
  __builtin_amdgcn_global_load_async_to_lds_b128(
      (__attribute__((address_space(1))) v4i*)g,
      (__attribute__((address_space(3))) v4i*)l, 0, 0);
}
__device__ __forceinline__ void wait_async0() {
#if __has_builtin(__builtin_amdgcn_s_wait_asynccnt)
  __builtin_amdgcn_s_wait_asynccnt(0);
#else
  asm volatile("s_wait_asynccnt 0" ::: "memory");
#endif
}
#endif

// A fragment (16x32 bf16, M=positions, K=channels) from f32 LDS row-major tile.
// lane<16: M=lane,    K = ks*32 + {0..7} ++ {16..23}
// lane>=16:M=lane-16, K = ks*32 + {8..15} ++ {24..31}
__device__ __forceinline__ v16bf load_a_frag(const float* lds, int row, int ks, int lane) {
  int kb = ks * 32 + ((lane >> 4) << 3);
  const float* rp = lds + row * AST + kb;
  v4f c0 = *(const v4f*)(rp);
  v4f c1 = *(const v4f*)(rp + 4);
  v4f c2 = *(const v4f*)(rp + 16);
  v4f c3 = *(const v4f*)(rp + 20);
  v16bf a;
#pragma unroll
  for (int z = 0; z < 4; ++z) {
    a[z]      = (__bf16)c0[z];
    a[4 + z]  = (__bf16)c1[z];
    a[8 + z]  = (__bf16)c2[z];
    a[12 + z] = (__bf16)c3[z];
  }
  return a;
}

// B fragment (32x16 bf16, K=in-ch, N=out-ch) from weight matrix w[oc][ic] (row-major 96x96).
// lane&15 = col (oc); lane<16 holds K = ks*32+{0..7,16..23}, lane>=16 holds K = ks*32+{8..15,24..31}.
__device__ __forceinline__ v16bf load_b_frag(const float* __restrict__ w, int ntBase, int ks, int lane) {
  int col = ntBase + (lane & 15);
  int kb  = ks * 32 + ((lane >> 4) << 3);
  const float* row = w + col * 96 + kb;
  v16bf bm;
#pragma unroll
  for (int i = 0; i < 16; ++i) bm[i] = (__bf16)row[i < 8 ? i : i + 8];
  return bm;
}

// ---------------------------------------------------------------------------
// Kernel 1: fused feat/value 1x1x1 conv (GEMM) + fold/head-split scatter.
// x: [4,96,32768] f32.  Output point-major: feat_ws/value_ws [bf=128][n=4096][24].
// grid = 1024 blocks (b * 256 chunks of 128 positions), 256 threads (8 waves).
// ---------------------------------------------------------------------------
__global__ __launch_bounds__(256) void fv_gemm_kernel(
    const float* __restrict__ x,
    const float* __restrict__ f_w, const float* __restrict__ f_b,
    const float* __restrict__ v_w, const float* __restrict__ v_b,
    float* __restrict__ feat_ws, float* __restrict__ value_ws) {
  __shared__ float xs[128 * AST];
  const int blk   = blockIdx.x;
  const int b     = blk >> 8;
  const int pbase = (blk & 255) << 7;
  const int t     = threadIdx.x;

  // stage x chunk [96 ic][128 pos] -> LDS f32 [pos][ic]: the transpose is done
  // by the per-lane LDS destination of the async B32 copy (or ds_store fallback).
#if USE_ASYNC_LDS
  for (int idx = t; idx < 96 * 128; idx += 256) {
    int ic = idx >> 7, j = idx & 127;
    async_copy_b32(x + (((size_t)b * 96 + ic) << 15) + pbase + j,
                   xs + j * AST + ic);
  }
  wait_async0();
  __syncthreads();
#else
  for (int idx = t; idx < 96 * 128; idx += 256) {
    int ic = idx >> 7, j = idx & 127;
    xs[j * AST + ic] = x[(((size_t)b * 96 + ic) << 15) + pbase + j];
  }
  __syncthreads();
#endif

  const int lane = t & 31, wv = t >> 5;
  v16bf a[3];
#pragma unroll
  for (int ks = 0; ks < 3; ++ks)
    a[ks] = load_a_frag(xs, wv * 16 + (lane & 15), ks, lane);

  const float* Wmat[2] = { f_w, v_w };
  const float* Bias[2] = { f_b, v_b };
  float*       Dst [2] = { feat_ws, value_ws };

#pragma unroll
  for (int arr = 0; arr < 2; ++arr) {
    const float* w = Wmat[arr];
    const float* bias = Bias[arr];
    float* dst = Dst[arr];
#pragma unroll
    for (int nt = 0; nt < 6; ++nt) {
      int oc = nt * 16 + (lane & 15);
      float bval = bias[oc];
      v8f acc;
#pragma unroll
      for (int r = 0; r < 8; ++r) acc[r] = bval;
#pragma unroll
      for (int ks = 0; ks < 3; ++ks) {
        v16bf bm = load_b_frag(w, nt * 16, ks, lane);
        acc = __builtin_amdgcn_wmma_f32_16x16x32_bf16(false, a[ks], false, bm,
                                                      (short)0, acc, false, false);
      }
      // scatter-store with head-split + fold folded into the address:
      // p -> (W,H,D); bf = (b*4+e)*8 + f1*4+f2*2+f3 ; n = (w,h,d) in 16^3
      int p0 = pbase + wv * 16 + ((lane >> 4) << 3);      // rows r add p0+r (same f3 block)
      int e = oc / 24, c24 = oc - e * 24;
      int Wi = p0 >> 10, Hi = (p0 >> 5) & 31, Di = p0 & 31;
      int bfi = ((b * 4 + e) << 3) | ((Wi >> 4) << 2) | ((Hi >> 4) << 1) | (Di >> 4);
      int n0  = ((Wi & 15) << 8) | ((Hi & 15) << 4) | (Di & 15);
      size_t base = ((size_t)bfi * 4096 + n0) * 24 + c24;
#pragma unroll
      for (int r = 0; r < 8; ++r) dst[base + (size_t)r * 24] = acc[r];
    }
  }
}

// ---------------------------------------------------------------------------
// Kernel 2: per sub-batch (bf) pooling -> cosine-sim assignment -> aggregate
// -> dispatch. One workgroup per bf (128 blocks), 256 threads.
// Writes point-major mid_ws [b=4][p=32768][96] (unfold + head-merge applied).
// ---------------------------------------------------------------------------
__global__ __launch_bounds__(256) void cluster_kernel(
    const float* __restrict__ feat_ws, const float* __restrict__ value_ws,
    const float* __restrict__ alpha_p, const float* __restrict__ beta_p,
    float* __restrict__ mid_ws) {
  __shared__ float cf[8][24], cv[8][24], cn[8][24], vc[8][24];
  __shared__ float aggN[8][24], aggOut[8][24], denom[8];
  __shared__ float sArr[4096];
  __shared__ unsigned char idxArr[4096];

  const int bf = blockIdx.x;
  const int t  = threadIdx.x;

  if (t < 192) { cf[t / 24][t % 24] = 0.f; cv[t / 24][t % 24] = 0.f; aggN[t / 24][t % 24] = 0.f; }
  if (t < 8) denom[t] = 0.f;
  __syncthreads();

  // ---- adaptive pooling (2x2x2 block means): 32 threads per cell, 16 pts each
  {
    int m = t >> 5, sub = t & 31;
    int pw = (m >> 2) & 1, ph = (m >> 1) & 1, pd = m & 1;
    float fsum[24], vsum[24];
#pragma unroll
    for (int k = 0; k < 24; ++k) { fsum[k] = 0.f; vsum[k] = 0.f; }
    for (int q = 0; q < 16; ++q) {
      int li = sub * 16 + q;
      int lw = li >> 6, lh = (li >> 3) & 7, ld = li & 7;
      int n = ((pw * 8 + lw) << 8) | ((ph * 8 + lh) << 4) | (pd * 8 + ld);
      size_t base = ((size_t)bf * 4096 + n) * 24;
      const v4f* fp = (const v4f*)(feat_ws + base);
      const v4f* vp = (const v4f*)(value_ws + base);
#pragma unroll
      for (int qq = 0; qq < 6; ++qq) {
        v4f a = fp[qq], c = vp[qq];
#pragma unroll
        for (int z = 0; z < 4; ++z) { fsum[qq * 4 + z] += a[z]; vsum[qq * 4 + z] += c[z]; }
      }
    }
#pragma unroll
    for (int k = 0; k < 24; ++k) {
      atomicAdd(&cf[m][k], fsum[k]);
      atomicAdd(&cv[m][k], vsum[k]);
    }
  }
  __syncthreads();

  if (t < 8) {
    float n2 = 0.f;
    for (int k = 0; k < 24; ++k) n2 += cf[t][k] * cf[t][k];
    float inv = 1.f / fmaxf(sqrtf(n2), 1e-12f);
    for (int k = 0; k < 24; ++k) {
      cn[t][k] = cf[t][k] * inv;            // normalized center direction
      vc[t][k] = cv[t][k] * (1.f / 512.f);  // value-center mean
    }
  }
  __syncthreads();

  const float alpha = alpha_p[0], beta = beta_p[0];

  // ---- assignment + aggregation
  for (int n = t; n < 4096; n += 256) {
    size_t base = ((size_t)bf * 4096 + n) * 24;
    float f[24];
    const v4f* fp = (const v4f*)(feat_ws + base);
#pragma unroll
    for (int qq = 0; qq < 6; ++qq) { v4f a = fp[qq];
#pragma unroll
      for (int z = 0; z < 4; ++z) f[qq * 4 + z] = a[z]; }
    float n2 = 0.f;
#pragma unroll
    for (int k = 0; k < 24; ++k) n2 += f[k] * f[k];
    float inv = 1.f / fmaxf(sqrtf(n2), 1e-12f);

    float bestS = -1.f; int bestM = 0;
#pragma unroll
    for (int m = 0; m < 8; ++m) {
      float dot = 0.f;
#pragma unroll
      for (int k = 0; k < 24; ++k) dot += cn[m][k] * f[k];
      float s = 1.f / (1.f + __expf(-(beta + alpha * dot * inv)));
      if (s > bestS) { bestS = s; bestM = m; }   // first-max wins (strict >)
    }
    sArr[n] = bestS; idxArr[n] = (unsigned char)bestM;

    const v4f* vp = (const v4f*)(value_ws + base);
    float v[24];
#pragma unroll
    for (int qq = 0; qq < 6; ++qq) { v4f a = vp[qq];
#pragma unroll
      for (int z = 0; z < 4; ++z) v[qq * 4 + z] = a[z]; }
#pragma unroll
    for (int k = 0; k < 24; ++k) atomicAdd(&aggN[bestM][k], bestS * v[k]);
    atomicAdd(&denom[bestM], bestS);
  }
  __syncthreads();

  if (t < 192) {
    int m = t / 24, k = t % 24;
    aggOut[m][k] = (aggN[m][k] + vc[m][k]) / (denom[m] + 1.f);
  }
  __syncthreads();

  // ---- dispatch back to points; write unfolded, head-merged layout [b][p][96]
  const int bb = bf >> 5;
  const int e  = (bf >> 3) & 3;
  const int f1 = (bf >> 2) & 1, f2 = (bf >> 1) & 1, f3 = bf & 1;
  for (int n = t; n < 4096; n += 256) {
    int wci = n >> 8, hci = (n >> 4) & 15, dci = n & 15;
    int p = ((f1 * 16 + wci) << 10) | ((f2 * 16 + hci) << 5) | (f3 * 16 + dci);
    int m = idxArr[n]; float s = sArr[n];
    v4f* d4 = (v4f*)(mid_ws + ((size_t)bb * 32768 + p) * 96 + e * 24);
#pragma unroll
    for (int qq = 0; qq < 6; ++qq) {
      v4f o;
#pragma unroll
      for (int z = 0; z < 4; ++z) o[z] = aggOut[m][qq * 4 + z] * s;
      d4[qq] = o;
    }
  }
}

// ---------------------------------------------------------------------------
// Kernel 3: projection GEMM. A = mid_ws [b][p][96] (contiguous rows), B = proj_w.
// Output final [b][96][32768] f32. grid = 1024 blocks, 256 threads (8 waves).
// ---------------------------------------------------------------------------
__global__ __launch_bounds__(256) void proj_gemm_kernel(
    const float* __restrict__ mid,
    const float* __restrict__ pw, const float* __restrict__ pb,
    float* __restrict__ out) {
  __shared__ float as[128 * AST];
  const int blk   = blockIdx.x;
  const int b     = blk >> 8;
  const int pbase = (blk & 255) << 7;
  const int t     = threadIdx.x;

  // stage mid chunk: contiguous 384B point rows -> async B128 copies into LDS
#if USE_ASYNC_LDS
  for (int i4 = t; i4 < 24 * 128; i4 += 256) {
    int j = i4 / 24, q = (i4 - j * 24) * 4;
    async_copy_b128(mid + ((size_t)b * 32768 + pbase + j) * 96 + q,
                    as + j * AST + q);
  }
  wait_async0();
  __syncthreads();
#else
  const v4f* m4 = (const v4f*)mid;
  for (int i4 = t; i4 < 24 * 128; i4 += 256) {
    int j = i4 / 24, q = i4 - j * 24;
    v4f val = m4[((size_t)b * 32768 + pbase + j) * 24 + q];
    *(v4f*)(as + j * AST + q * 4) = val;
  }
  __syncthreads();
#endif

  const int lane = t & 31, wv = t >> 5;
  v16bf a[3];
#pragma unroll
  for (int ks = 0; ks < 3; ++ks)
    a[ks] = load_a_frag(as, wv * 16 + (lane & 15), ks, lane);

#pragma unroll
  for (int nt = 0; nt < 6; ++nt) {
    int oc = nt * 16 + (lane & 15);
    float bval = pb[oc];
    v8f acc;
#pragma unroll
    for (int r = 0; r < 8; ++r) acc[r] = bval;
#pragma unroll
    for (int ks = 0; ks < 3; ++ks) {
      v16bf bm = load_b_frag(pw, nt * 16, ks, lane);
      acc = __builtin_amdgcn_wmma_f32_16x16x32_bf16(false, a[ks], false, bm,
                                                    (short)0, acc, false, false);
    }
    int p0 = pbase + wv * 16 + ((lane >> 4) << 3);
    float* dst = out + (((size_t)b * 96 + oc) << 15) + p0;
    v4f lo, hi;
#pragma unroll
    for (int z = 0; z < 4; ++z) { lo[z] = acc[z]; hi[z] = acc[4 + z]; }
    *(v4f*)dst = lo;
    *(v4f*)(dst + 4) = hi;
  }
}

// ---------------------------------------------------------------------------
extern "C" void kernel_launch(void* const* d_in, const int* in_sizes, int n_in,
                              void* d_out, int out_size, void* d_ws, size_t ws_size,
                              hipStream_t stream) {
  (void)in_sizes; (void)n_in; (void)out_size; (void)ws_size;
  const float* x      = (const float*)d_in[0];
  const float* f_w    = (const float*)d_in[1];
  const float* f_b    = (const float*)d_in[2];
  const float* v_w    = (const float*)d_in[3];
  const float* v_b    = (const float*)d_in[4];
  const float* proj_w = (const float*)d_in[5];
  const float* proj_b = (const float*)d_in[6];
  const float* alpha  = (const float*)d_in[7];
  const float* beta   = (const float*)d_in[8];
  float* out = (float*)d_out;

  const size_t SUB = (size_t)128 * 4096 * 24;   // 12.58M floats per array
  float* feat_ws  = (float*)d_ws;
  float* value_ws = feat_ws + SUB;
  float* mid_ws   = value_ws + SUB;

  fv_gemm_kernel<<<dim3(1024), dim3(256), 0, stream>>>(x, f_w, f_b, v_w, v_b,
                                                       feat_ws, value_ws);
  cluster_kernel<<<dim3(128), dim3(256), 0, stream>>>(feat_ws, value_ws,
                                                      alpha, beta, mid_ws);
  proj_gemm_kernel<<<dim3(1024), dim3(256), 0, stream>>>(mid_ws, proj_w, proj_b, out);
}
